// GATLayer_1735166787777
// MI455X (gfx1250) — matrix operand order repaired and checked by reading, not compiled, not run on hardware
//
#include <hip/hip_runtime.h>
#include <math.h>

typedef __attribute__((ext_vector_type(16))) _Float16 v16h;
typedef __attribute__((ext_vector_type(8)))  float    v8f;

#define N_NODES   50000
#define N_EDGES   1600000
#define IN_DIM    128
#define OUT_DIM   128
#define NUM_HEADS 4
#define HEAD_DIM  32
#define INV_SQRT_HD 0.17677669529663687f  // 1/sqrt(32)

// ---------- order-preserving float<->uint key for atomicMax on floats ----------
__device__ __forceinline__ unsigned fmax_key(float x) {
    unsigned b = __float_as_uint(x);
    return (b & 0x80000000u) ? ~b : (b | 0x80000000u);
}
__device__ __forceinline__ float fmax_unkey(unsigned k) {
    unsigned b = (k & 0x80000000u) ? (k ^ 0x80000000u) : ~k;
    return __uint_as_float(b);
}

// ---------------------------------------------------------------------------
// 16x16 output tile of  D = A(16x128,f32 rows) * B(16x128,f32 rows)^T  using
// v_wmma_f32_16x16x32_f16, K swept 128 in 4 steps. Layouts per CDNA5 ISA:
//   A frag: M = lane%16, K = (e/8)*16 + (lane/16)*8 + e%8
//   B frag: N = lane%16, K = (e/8)*16 + (lane/16)*8 + e%8
// arow = row (lane%16) of A-tile, wrow = row (lane%16) of W (B = W^T).
// ---------------------------------------------------------------------------
__device__ __forceinline__ v8f tile_gemm_k128(const float* __restrict__ arow,
                                              const float* __restrict__ wrow,
                                              int hi) {
    v8f acc = {};
#pragma unroll
    for (int kk = 0; kk < 128; kk += 32) {
        const float4 a0 = *reinterpret_cast<const float4*>(arow + kk + hi * 8);
        const float4 a1 = *reinterpret_cast<const float4*>(arow + kk + hi * 8 + 4);
        const float4 a2 = *reinterpret_cast<const float4*>(arow + kk + hi * 8 + 16);
        const float4 a3 = *reinterpret_cast<const float4*>(arow + kk + hi * 8 + 20);
        const float4 b0 = *reinterpret_cast<const float4*>(wrow + kk + hi * 8);
        const float4 b1 = *reinterpret_cast<const float4*>(wrow + kk + hi * 8 + 4);
        const float4 b2 = *reinterpret_cast<const float4*>(wrow + kk + hi * 8 + 16);
        const float4 b3 = *reinterpret_cast<const float4*>(wrow + kk + hi * 8 + 20);
        v16h a, b;
        a[0]=(_Float16)a0.x; a[1]=(_Float16)a0.y; a[2]=(_Float16)a0.z; a[3]=(_Float16)a0.w;
        a[4]=(_Float16)a1.x; a[5]=(_Float16)a1.y; a[6]=(_Float16)a1.z; a[7]=(_Float16)a1.w;
        a[8]=(_Float16)a2.x; a[9]=(_Float16)a2.y; a[10]=(_Float16)a2.z; a[11]=(_Float16)a2.w;
        a[12]=(_Float16)a3.x; a[13]=(_Float16)a3.y; a[14]=(_Float16)a3.z; a[15]=(_Float16)a3.w;
        b[0]=(_Float16)b0.x; b[1]=(_Float16)b0.y; b[2]=(_Float16)b0.z; b[3]=(_Float16)b0.w;
        b[4]=(_Float16)b1.x; b[5]=(_Float16)b1.y; b[6]=(_Float16)b1.z; b[7]=(_Float16)b1.w;
        b[8]=(_Float16)b2.x; b[9]=(_Float16)b2.y; b[10]=(_Float16)b2.z; b[11]=(_Float16)b2.w;
        b[12]=(_Float16)b3.x; b[13]=(_Float16)b3.y; b[14]=(_Float16)b3.z; b[15]=(_Float16)b3.w;
        acc = __builtin_amdgcn_wmma_f32_16x16x32_f16(false, a, false, b,
                                                     (short)0, acc, false, false);
    }
    return acc;
}

// ------------------------- QKV projections (fused) -------------------------
// grid = (N/16, 3); block = 256 (8 waves). Wave w computes cols [16w,16w+16)
// of projection blockIdx.y for node rows [16*bx, 16*bx+16).
__global__ __launch_bounds__(256)
void gat_gemm_qkv(const float* __restrict__ x,
                  const float* __restrict__ Wq, const float* __restrict__ bq,
                  const float* __restrict__ Wk, const float* __restrict__ bk,
                  const float* __restrict__ Wv, const float* __restrict__ bv,
                  float* __restrict__ Q, float* __restrict__ K,
                  float* __restrict__ V) {
    const int lane = threadIdx.x & 31;
    const int wave = threadIdx.x >> 5;
    const int hi = lane >> 4;
    const int lo = lane & 15;
    const float* W; const float* bias; float* out;
    if (blockIdx.y == 0)      { W = Wq; bias = bq; out = Q; }
    else if (blockIdx.y == 1) { W = Wk; bias = bk; out = K; }
    else                      { W = Wv; bias = bv; out = V; }
    const int rowBase = blockIdx.x * 16;
    const int n0 = wave * 16;
    const float* arow = x + (size_t)(rowBase + lo) * IN_DIM;
    const float* wrow = W + (size_t)(n0 + lo) * IN_DIM;
    v8f acc = tile_gemm_k128(arow, wrow, hi);
    const float bn = bias[n0 + lo];
#pragma unroll
    for (int j = 0; j < 8; ++j) {
        const int m = rowBase + j + hi * 8;          // D: M = j + 8*(lane/16)
        out[(size_t)m * OUT_DIM + n0 + lo] = acc[j] + bn;  // D: N = lane%16
    }
}

// ------------------------- workspace init ----------------------------------
__global__ __launch_bounds__(256)
void gat_init(float* __restrict__ zbuf, size_t n, unsigned* __restrict__ gmax) {
    size_t i = (size_t)blockIdx.x * blockDim.x + threadIdx.x;
    const size_t stride = (size_t)gridDim.x * blockDim.x;
    for (; i < n; i += stride) zbuf[i] = 0.0f;
    if (blockIdx.x == 0 && threadIdx.x == 0) *gmax = 0u;  // below any finite key
}

// per-edge, per-lane: head h = lane/8, dims [4*lane, 4*lane+4). Returns the
// head score (broadcast to all 8 lanes of the head group via xor-reduction).
__device__ __forceinline__ float edge_head_score(const float* __restrict__ Q,
                                                 const float* __restrict__ Kf,
                                                 int src, int dst, int lane) {
    const float4 q = *reinterpret_cast<const float4*>(Q + (size_t)dst * OUT_DIM + lane * 4);
    const float4 k = *reinterpret_cast<const float4*>(Kf + (size_t)src * OUT_DIM + lane * 4);
    float p = q.x * k.x + q.y * k.y + q.z * k.z + q.w * k.w;
    p += __shfl_xor(p, 1, 32);
    p += __shfl_xor(p, 2, 32);
    p += __shfl_xor(p, 4, 32);
    return p * INV_SQRT_HD;
}

// ------------------------- pass 1: global score max ------------------------
__global__ __launch_bounds__(256)
void gat_edge_max(const float* __restrict__ Q, const float* __restrict__ Kf,
                  const int* __restrict__ ei, unsigned* __restrict__ gmax) {
    __shared__ float smax[8];
    const int lane = threadIdx.x & 31;
    const int wave = threadIdx.x >> 5;
    const int wid = (blockIdx.x * blockDim.x + threadIdx.x) >> 5;
    const int nwaves = (gridDim.x * blockDim.x) >> 5;
    float lmax = -3.4e38f;
    for (int e = wid; e < N_EDGES; e += nwaves) {
        const int src = ei[e];
        const int dst = ei[N_EDGES + e];
        lmax = fmaxf(lmax, edge_head_score(Q, Kf, src, dst, lane));
    }
    lmax = fmaxf(lmax, __shfl_xor(lmax, 8, 32));
    lmax = fmaxf(lmax, __shfl_xor(lmax, 16, 32));
    if (lane == 0) smax[wave] = lmax;
    __syncthreads();
    if (threadIdx.x == 0) {
        float m = smax[0];
#pragma unroll
        for (int i = 1; i < 8; ++i) m = fmaxf(m, smax[i]);
        atomicMax(gmax, fmax_key(m));
    }
}

// ------------------- pass 2: exp + per-dst segment sums --------------------
__global__ __launch_bounds__(256)
void gat_edge_expsum(const float* __restrict__ Q, const float* __restrict__ Kf,
                     const int* __restrict__ ei, const unsigned* __restrict__ gmax,
                     float* __restrict__ sums) {
    const int lane = threadIdx.x & 31;
    const int wid = (blockIdx.x * blockDim.x + threadIdx.x) >> 5;
    const int nwaves = (gridDim.x * blockDim.x) >> 5;
    const float gm = fmax_unkey(*gmax);
    for (int e = wid; e < N_EDGES; e += nwaves) {
        const int src = ei[e];
        const int dst = ei[N_EDGES + e];
        const float s = edge_head_score(Q, Kf, src, dst, lane);
        const float w = __expf(s - gm);
        if ((lane & 7) == 0)
            atomicAdd(&sums[(size_t)dst * NUM_HEADS + (lane >> 3)], w);
    }
}

// ------------- pass 3: attention-weighted scatter of V ---------------------
__global__ __launch_bounds__(256)
void gat_edge_aggregate(const float* __restrict__ Q, const float* __restrict__ Kf,
                        const float* __restrict__ V, const int* __restrict__ ei,
                        const unsigned* __restrict__ gmax,
                        const float* __restrict__ sums, float* __restrict__ agg) {
    const int lane = threadIdx.x & 31;
    const int wid = (blockIdx.x * blockDim.x + threadIdx.x) >> 5;
    const int nwaves = (gridDim.x * blockDim.x) >> 5;
    const float gm = fmax_unkey(*gmax);
    for (int e = wid; e < N_EDGES; e += nwaves) {
        const int src = ei[e];
        const int dst = ei[N_EDGES + e];
        const float s = edge_head_score(Q, Kf, src, dst, lane);   // same math as pass 2
        const float w = __expf(s - gm);
        const float att = w / (sums[(size_t)dst * NUM_HEADS + (lane >> 3)] + 1e-8f);
        const float4 v = *reinterpret_cast<const float4*>(V + (size_t)src * OUT_DIM + lane * 4);
        float* a = agg + (size_t)dst * OUT_DIM + lane * 4;
        atomicAdd(a + 0, v.x * att);
        atomicAdd(a + 1, v.y * att);
        atomicAdd(a + 2, v.z * att);
        atomicAdd(a + 3, v.w * att);
    }
}

// ------------------------- output projection + ReLU ------------------------
__global__ __launch_bounds__(256)
void gat_gemm_out(const float* __restrict__ agg, const float* __restrict__ Wo,
                  const float* __restrict__ bo, float* __restrict__ out) {
    const int lane = threadIdx.x & 31;
    const int wave = threadIdx.x >> 5;
    const int hi = lane >> 4;
    const int lo = lane & 15;
    const int rowBase = blockIdx.x * 16;
    const int n0 = wave * 16;
    const float* arow = agg + (size_t)(rowBase + lo) * OUT_DIM;
    const float* wrow = Wo + (size_t)(n0 + lo) * OUT_DIM;
    v8f acc = tile_gemm_k128(arow, wrow, hi);
    const float bn = bo[n0 + lo];
#pragma unroll
    for (int j = 0; j < 8; ++j) {
        const int m = rowBase + j + hi * 8;
        out[(size_t)m * OUT_DIM + n0 + lo] = fmaxf(acc[j] + bn, 0.0f);
    }
}

extern "C" void kernel_launch(void* const* d_in, const int* in_sizes, int n_in,
                              void* d_out, int out_size, void* d_ws, size_t ws_size,
                              hipStream_t stream) {
    const float* x  = (const float*)d_in[0];
    const int*   ei = (const int*)d_in[1];
    const float* Wq = (const float*)d_in[2]; const float* bq = (const float*)d_in[3];
    const float* Wk = (const float*)d_in[4]; const float* bk = (const float*)d_in[5];
    const float* Wv = (const float*)d_in[6]; const float* bv = (const float*)d_in[7];
    const float* Wo = (const float*)d_in[8]; const float* bo = (const float*)d_in[9];
    float* out = (float*)d_out;

    // workspace layout (floats): Q | K | V | agg | sums | gmax   (~103.2 MB)
    const size_t NE = (size_t)N_NODES * OUT_DIM;       // 6,400,000
    float* Q    = (float*)d_ws;
    float* Kf   = Q + NE;
    float* V    = Kf + NE;
    float* agg  = V + NE;
    float* sums = agg + NE;                            // N_NODES*NUM_HEADS
    unsigned* gmax = (unsigned*)(sums + (size_t)N_NODES * NUM_HEADS);

    // 1) zero agg+sums (contiguous) and reset global max key
    gat_init<<<dim3(2048), 256, 0, stream>>>(agg, NE + (size_t)N_NODES * NUM_HEADS, gmax);
    // 2) Q/K/V projections via WMMA
    gat_gemm_qkv<<<dim3(N_NODES / 16, 3), 256, 0, stream>>>(x, Wq, bq, Wk, bk, Wv, bv, Q, Kf, V);
    // 3) global score max
    gat_edge_max<<<dim3(4096), 256, 0, stream>>>(Q, Kf, ei, gmax);
    // 4) exp + per-destination sums
    gat_edge_expsum<<<dim3(8192), 256, 0, stream>>>(Q, Kf, ei, gmax, sums);
    // 5) attention-weighted aggregation of V
    gat_edge_aggregate<<<dim3(8192), 256, 0, stream>>>(Q, Kf, V, ei, gmax, sums, agg);
    // 6) output projection + ReLU via WMMA
    gat_gemm_out<<<dim3(N_NODES / 16, 1), 256, 0, stream>>>(agg, Wo, bo, out);
}
